// MultiHeadSelfAttention_22789096472938
// MI455X (gfx1250) — matrix operand order repaired and checked by reading, not compiled, not run on hardware
//
#include <hip/hip_runtime.h>
#include <hip/hip_bf16.h>

#define DIMN 1024
#define SEQ  2048
#define BS   4096   // 2 * 2048
#define DH   64

typedef __attribute__((ext_vector_type(16))) __bf16 v16bf;
typedef __attribute__((ext_vector_type(8)))  float  v8f;

struct U44 { uint4 a, b; };
struct U8  { unsigned u[8]; };

__device__ __forceinline__ unsigned short f2bf(float f) {
  unsigned u = __builtin_bit_cast(unsigned, f);
  u += 0x7FFFu + ((u >> 16) & 1u);      // round-to-nearest-even
  return (unsigned short)(u >> 16);
}
__device__ __forceinline__ unsigned pk2(float lo, float hi) {
  return (unsigned)f2bf(lo) | ((unsigned)f2bf(hi) << 16);
}
// bf16 fragment from two 16-byte chunks (K = half*16+e, contiguous k order)
__device__ __forceinline__ v16bf frag2(const unsigned short* p0, const unsigned short* p1) {
  U44 t; t.a = *(const uint4*)p0; t.b = *(const uint4*)p1;
  return __builtin_bit_cast(v16bf, t);
}
__device__ __forceinline__ v8f wmma_bf16(v16bf a, v16bf b, v8f c) {
  return __builtin_amdgcn_wmma_f32_16x16x32_bf16(false, a, false, b, (short)0, c, false, false);
}
__device__ __forceinline__ float rmax16(float v) {
  #pragma unroll
  for (int m = 1; m < 16; m <<= 1) v = fmaxf(v, __shfl_xor(v, m, 16));
  return v;
}

// ---------------- one-shot f32 -> bf16 convert (optionally scaled) ----------------
__global__ __launch_bounds__(256) void cvt_bf16(const float* __restrict__ in,
                                                unsigned short* __restrict__ out,
                                                int n8, float scl) {
  int i = blockIdx.x * blockDim.x + threadIdx.x;
  if (i >= n8) return;
  const float4* p = (const float4*)in + (size_t)i * 2;
  float4 a = p[0], b = p[1];
  uint4 r;
  r.x = pk2(a.x * scl, a.y * scl);
  r.y = pk2(a.z * scl, a.w * scl);
  r.z = pk2(b.x * scl, b.y * scl);
  r.w = pk2(b.z * scl, b.w * scl);
  ((uint4*)out)[i] = r;
}

// ---------------- QKV projection: y = x @ W^T + b, all-bf16 inner loop ----------------
// grid (512, 3), block 128. wave -> 32x64 output tile. V stored transposed [B][DIM][S].
// Wq/bq pre-scaled by 1/sqrt(DH) so attention scores need no scaling.
__global__ __launch_bounds__(128) void qkv_proj(
    const unsigned short* __restrict__ xb,
    const unsigned short* __restrict__ Wqb, const float* __restrict__ bq,
    const unsigned short* __restrict__ Wkb, const float* __restrict__ bk,
    const unsigned short* __restrict__ Wvb, const float* __restrict__ bv,
    unsigned short* __restrict__ Q, unsigned short* __restrict__ K,
    unsigned short* __restrict__ Vt)
{
  const int lane = threadIdx.x & 31;
  const int wave = blockIdx.x * 4 + (threadIdx.x >> 5);
  const int mat  = blockIdx.y;            // 0=Q 1=K 2=V
  const int trow = (wave >> 4) * 32;      // 32-row tile
  const int ncol = (wave & 15) * 64;      // 64-col group
  const int n    = lane & 15;
  const int half = lane >> 4;

  const unsigned short* W = (mat == 0) ? Wqb : (mat == 1) ? Wkb : Wvb;
  const float* bias       = (mat == 0) ? bq  : (mat == 1) ? bk  : bv;
  const float bscl        = (mat == 0) ? 0.125f : 1.0f;

  v8f acc[2][4] = {};
  for (int kb = 0; kb < DIMN; kb += 32) {
    v16bf a[2];
    #pragma unroll
    for (int r = 0; r < 2; ++r) {
      const unsigned short* xr = xb + (size_t)(trow + r*16 + n) * DIMN + kb;
      a[r] = frag2(xr + half * 16, xr + half * 16 + 8);
    }
    #pragma unroll
    for (int c = 0; c < 4; ++c) {
      const unsigned short* wr = W + (size_t)(ncol + c*16 + n) * DIMN + kb + half * 16;
      v16bf b = frag2(wr, wr + 8);
      acc[0][c] = wmma_bf16(a[0], b, acc[0][c]);
      acc[1][c] = wmma_bf16(a[1], b, acc[1][c]);
    }
  }

  #pragma unroll
  for (int r = 0; r < 2; ++r) {
    if (mat < 2) {
      unsigned short* out = (mat == 0) ? Q : K;
      #pragma unroll
      for (int c = 0; c < 4; ++c) {
        float bc = bias[ncol + c*16 + n] * bscl;
        #pragma unroll
        for (int j = 0; j < 8; ++j)
          out[(size_t)(trow + r*16 + half*8 + j) * DIMN + ncol + c*16 + n] =
              f2bf(acc[r][c][j] + bc);
      }
    } else {
      #pragma unroll
      for (int c = 0; c < 4; ++c) {
        int d = ncol + c*16 + n;
        float bc = bias[d];
        #pragma unroll
        for (int j = 0; j < 8; ++j) {
          int t = trow + r*16 + half*8 + j;
          Vt[((size_t)(t >> 11) * DIMN + d) * SEQ + (t & 2047)] = f2bf(acc[r][c][j] + bc);
        }
      }
    }
  }
}

// ---------------- flash attention: one wave = one 16-query tile of one (b,h) ----------------
__global__ __launch_bounds__(128) void attn(
    const unsigned short* __restrict__ Q,
    const unsigned short* __restrict__ K,
    const unsigned short* __restrict__ Vt,
    unsigned short* __restrict__ ctx)
{
  __shared__ __align__(16) unsigned short plds[4][16 * 32];
  const int lane = threadIdx.x & 31;
  const int widx = threadIdx.x >> 5;
  const int wave = blockIdx.x * 4 + widx;
  const int bh   = wave >> 7;           // 0..31
  const int qt   = wave & 127;
  const int b    = bh >> 4;
  const int h    = bh & 15;
  const int n    = lane & 15;
  const int half = lane >> 4;
  const int qrow = b * SEQ + qt * 16;
  unsigned short* my = &plds[widx][0];

  // Q fragments (already scaled by 1/sqrt(DH) at projection time)
  v16bf aq[2];
  #pragma unroll
  for (int d = 0; d < 2; ++d) {
    const unsigned short* qr = Q + (size_t)(qrow + n) * DIMN + h * DH + d * 32;
    aq[d] = frag2(qr + half * 16, qr + half * 16 + 8);
  }
  // all-ones B fragment: rowsum-by-WMMA
  U8 uo;
  #pragma unroll
  for (int i = 0; i < 8; ++i) uo.u[i] = 0x3F803F80u;
  const v16bf ones = __builtin_bit_cast(v16bf, uo);

  v8f acc[4] = {};
  v8f accl = {};                       // flash "l" state, one WMMA per block
  float rowmax[8];
  #pragma unroll
  for (int j = 0; j < 8; ++j) rowmax[j] = -1e30f;

  for (int kb = 0; kb < SEQ; kb += 32) {
    // scores: two 16-key N-tiles, K dim = Dh (2 x 32)
    v8f s[2];
    #pragma unroll
    for (int nt = 0; nt < 2; ++nt) {
      v8f sc = {};
      #pragma unroll
      for (int d = 0; d < 2; ++d) {
        const unsigned short* kr =
            K + (size_t)(b*SEQ + kb + nt*16 + n) * DIMN + h * DH + d*32 + half * 16;
        sc = wmma_bf16(aq[d], frag2(kr, kr + 8), sc);
      }
      s[nt] = sc;
    }
    // online max per row; row sums are delegated to the ones-WMMA below
    float f[8];
    #pragma unroll
    for (int j = 0; j < 8; ++j) {
      float mt   = rmax16(fmaxf(s[0][j], s[1][j]));
      float mnew = fmaxf(rowmax[j], mt);
      f[j]       = __expf(rowmax[j] - mnew);
      rowmax[j]  = mnew;
      float p0 = __expf(s[0][j] - mnew), p1 = __expf(s[1][j] - mnew);
      my[(half*8 + j) * 32 + n]      = f2bf(p0);   // P -> LDS (row-major 16x32)
      my[(half*8 + j) * 32 + 16 + n] = f2bf(p1);
    }
    #pragma unroll
    for (int j = 0; j < 8; ++j) {
      accl[j] *= f[j];
      #pragma unroll
      for (int c = 0; c < 4; ++c) acc[c][j] *= f[j];
    }
    asm volatile("s_wait_dscnt 0" ::: "memory");   // LDS same-wave is in order; wait returns
    const unsigned short* pr = my + n * 32;        // re-read P in A layout
    v16bf ap = frag2(pr + half * 16, pr + half * 16 + 8);
    accl = wmma_bf16(ap, ones, accl);              // l += row-sums of P (matrix pipe)
    #pragma unroll
    for (int c = 0; c < 4; ++c) {
      const unsigned short* vr =
          Vt + ((size_t)b * DIMN + h * DH + c*16 + n) * SEQ + kb + half * 16;
      acc[c] = wmma_bf16(ap, frag2(vr, vr + 8), acc[c]);
    }
  }

  float inv[8];
  #pragma unroll
  for (int j = 0; j < 8; ++j) inv[j] = 1.0f / accl[j];
  #pragma unroll
  for (int c = 0; c < 4; ++c)
    #pragma unroll
    for (int j = 0; j < 8; ++j)
      ctx[(size_t)(qrow + half*8 + j) * DIMN + h * DH + c*16 + n] = f2bf(acc[c][j] * inv[j]);
}

// ---------------- output projection: out = ctx @ Wo^T + bo (f32 out) ----------------
// grid 512, block 128. wave -> 32x64 tile, all-bf16 inner loop.
__global__ __launch_bounds__(128) void out_proj(
    const unsigned short* __restrict__ ctx,
    const unsigned short* __restrict__ Wob, const float* __restrict__ bo,
    float* __restrict__ out)
{
  const int lane = threadIdx.x & 31;
  const int wave = blockIdx.x * 4 + (threadIdx.x >> 5);
  const int trow = (wave >> 4) * 32;
  const int ncol = (wave & 15) * 64;
  const int n    = lane & 15;
  const int half = lane >> 4;

  v8f acc[2][4] = {};
  for (int kb = 0; kb < DIMN; kb += 32) {
    v16bf a[2];
    #pragma unroll
    for (int r = 0; r < 2; ++r) {
      const unsigned short* cr = ctx + (size_t)(trow + r*16 + n) * DIMN + kb;
      a[r] = frag2(cr + half * 16, cr + half * 16 + 8);
    }
    #pragma unroll
    for (int c = 0; c < 4; ++c) {
      const unsigned short* wr = Wob + (size_t)(ncol + c*16 + n) * DIMN + kb + half * 16;
      v16bf b = frag2(wr, wr + 8);
      acc[0][c] = wmma_bf16(a[0], b, acc[0][c]);
      acc[1][c] = wmma_bf16(a[1], b, acc[1][c]);
    }
  }
  #pragma unroll
  for (int r = 0; r < 2; ++r)
    #pragma unroll
    for (int c = 0; c < 4; ++c) {
      float bc = bo[ncol + c*16 + n];
      #pragma unroll
      for (int j = 0; j < 8; ++j)
        out[(size_t)(trow + r*16 + half*8 + j) * DIMN + ncol + c*16 + n] = acc[r][c][j] + bc;
    }
}

extern "C" void kernel_launch(void* const* d_in, const int* in_sizes, int n_in,
                              void* d_out, int out_size, void* d_ws, size_t ws_size,
                              hipStream_t stream) {
  (void)in_sizes; (void)n_in; (void)out_size; (void)ws_size;
  const float* x  = (const float*)d_in[0];
  const float* Wq = (const float*)d_in[1];
  const float* bq = (const float*)d_in[2];
  const float* Wk = (const float*)d_in[3];
  const float* bk = (const float*)d_in[4];
  const float* Wv = (const float*)d_in[5];
  const float* bv = (const float*)d_in[6];
  const float* Wo = (const float*)d_in[7];
  const float* bo = (const float*)d_in[8];
  float* out = (float*)d_out;

  const size_t NX = (size_t)BS * DIMN;      // 4M elems
  const size_t NW = (size_t)DIMN * DIMN;    // 1M elems
  unsigned short* xb  = (unsigned short*)d_ws;   // bf16 x        (8 MB)
  unsigned short* Wqb = xb  + NX;                // bf16 Wq/8     (2 MB)
  unsigned short* Wkb = Wqb + NW;
  unsigned short* Wvb = Wkb + NW;
  unsigned short* Wob = Wvb + NW;
  unsigned short* Qb  = Wob + NW;                // [BS][DIM]     (8 MB)
  unsigned short* Kb  = Qb  + NX;                // [BS][DIM]     (8 MB)
  unsigned short* Vt  = Kb  + NX;                // [B][DIM][S]   (8 MB)
  unsigned short* Cx  = Vt  + NX;                // [BS][DIM]     (8 MB)

  dim3 blk(128), blc(256);
  const float qscale = 0.125f;                   // 1/sqrt(64)
  cvt_bf16<<<dim3((int)(NX / 8 / 256)), blc, 0, stream>>>(x,  xb,  (int)(NX / 8), 1.0f);
  cvt_bf16<<<dim3((int)(NW / 8 / 256)), blc, 0, stream>>>(Wq, Wqb, (int)(NW / 8), qscale);
  cvt_bf16<<<dim3((int)(NW / 8 / 256)), blc, 0, stream>>>(Wk, Wkb, (int)(NW / 8), 1.0f);
  cvt_bf16<<<dim3((int)(NW / 8 / 256)), blc, 0, stream>>>(Wv, Wvb, (int)(NW / 8), 1.0f);
  cvt_bf16<<<dim3((int)(NW / 8 / 256)), blc, 0, stream>>>(Wo, Wob, (int)(NW / 8), 1.0f);

  qkv_proj<<<dim3(512, 3), blk, 0, stream>>>(xb, Wqb, bq, Wkb, bk, Wvb, bv, Qb, Kb, Vt);
  attn    <<<dim3(1024),   blk, 0, stream>>>(Qb, Kb, Vt, Cx);
  out_proj<<<dim3(512),    blk, 0, stream>>>(Cx, Wob, bo, out);
}
